// GambaBlock_7000796692742
// MI455X (gfx1250) — compile-verified, matrix operand order checked by reading, not run on hardware
//
#include <hip/hip_runtime.h>

typedef __attribute__((ext_vector_type(16))) __bf16 bf16x16;
typedef __attribute__((ext_vector_type(8)))  float  f32x8;
typedef __attribute__((ext_vector_type(4)))  unsigned int u32x4;
typedef __attribute__((ext_vector_type(8)))  unsigned int u32x8;

// ---------------------------------------------------------------------------
// bf16-WMMA GEMM with TDM (tensor_load_to_lds) double-buffered B panels.
//   C = act( remap(A) @ B + bias ),  A: MxK fp32 (row stride LDA),
//   B: KxN fp32 row-major.  Row remap: r -> (r/rpb)*obs + off + r%rpb.
// Block = 256 threads = 8 waves; wave strip = 16(M) x 64(N) (4 accumulators);
// block tile = 128 x 64.  B panel = 32x64 f32 tile DMA'd into LDS by wave 0
// via the Tensor Data Mover; completion tracked with TENSORcnt.
// Launcher guarantees M%128==0, N%64==0, K%32==0.
// ---------------------------------------------------------------------------
template<int N, int K, int LDA, bool SOFTPLUS>
__global__ __launch_bounds__(256) void gamba_gemm(
    const float* __restrict__ A,
    const float* __restrict__ B,
    const float* __restrict__ bias,
    float* __restrict__ C,
    int a_rpb, long long a_obs, long long a_off,
    int c_rpb, long long c_obs, long long c_off)
{
    __shared__ float Bs[2][32 * 64];        // two 8 KB panels, [kk*64 + n]

    const int t    = threadIdx.x;
    const int lane = t & 31;
    const int wave = t >> 5;
    const int l15  = lane & 15;
    const int hi   = lane >> 4;

    const int rowBase = blockIdx.y * 128 + wave * 16;
    const int colBase = blockIdx.x * 64;

    // A fragment rows: 16-bit A layout -> lane%16 == M
    const int m = rowBase + l15;
    const long long arow = (long long)(m / a_rpb) * a_obs + a_off + (m % a_rpb);
    const float* __restrict__ Arow = A + arow * (long long)LDA;

    // Tensor-DMA descriptor issue (wave 0 only; TDM ignores EXEC, a scalar
    // branch skips issue entirely).  2-group D# => 2D tensor / 2D tile.
    const unsigned lds0 = (unsigned)(uintptr_t)&Bs[0][0]; // low32 == LDS offset
    const unsigned lds1 = (unsigned)(uintptr_t)&Bs[1][0];

    auto tdm_issue = [&](int k0, unsigned ldsAddr) {
        const unsigned long long ga =
            (unsigned long long)(uintptr_t)(B + (long long)k0 * N + colBase);
        u32x4 g0;
        g0[0] = 1u;                                   // count=1, no gather
        g0[1] = ldsAddr;                              // lds_addr (bytes)
        g0[2] = (unsigned)ga;                         // global_addr[31:0]
        g0[3] = (unsigned)(ga >> 32) | (2u << 30);    // ga[56:32] | type=2
        u32x8 g1;
        g1[0] = 2u << 16;                             // data_size = 4 bytes
        g1[1] = ((unsigned)N & 0xFFFFu) << 16;        // tensor_dim0[15:0]
        g1[2] = ((unsigned)N >> 16) | (((unsigned)K & 0xFFFFu) << 16); // d0 hi | d1 lo
        g1[3] = ((unsigned)K >> 16) | (64u << 16);    // d1 hi | tile_dim0=64
        g1[4] = 32u;                                  // tile_dim1=32, tile_dim2=0
        g1[5] = (unsigned)N;                          // tensor_dim0_stride
        g1[6] = 0u;
        g1[7] = 0u;
        asm volatile("tensor_load_to_lds %0, %1" : : "s"(g0), "s"(g1) : "memory");
    };

    constexpr int STEPS = K / 32;
    if (wave == 0) tdm_issue(0, lds0);                // prologue: panel 0 in flight

    f32x8 acc[4] = {};

    for (int i = 0; i < STEPS; ++i) {
        const int k0 = i * 32;
        if (wave == 0) {
            if (i + 1 < STEPS) {                      // overlap next DMA
                tdm_issue(k0 + 32, (i & 1) ? lds0 : lds1);
                __builtin_amdgcn_s_wait_tensorcnt(1); // current panel done
            } else {
                __builtin_amdgcn_s_wait_tensorcnt(0);
            }
        }
        __syncthreads();                              // panel visible to all waves

        const float* __restrict__ P = &Bs[i & 1][0] + l15;

        // ---- A fragment: 8 contiguous f32 pairs -> packed bf16 ----
        bf16x16 af;
#pragma unroll
        for (int v = 0; v < 8; ++v) {
            const int k = k0 + ((v < 4) ? 0 : 16) + hi * 8 + (v & 3) * 2;
            const float2 a2 = *(const float2*)&Arow[k];   // global_load_b64
            af[2 * v]     = (__bf16)a2.x;
            af[2 * v + 1] = (__bf16)a2.y;
        }

        // ---- 4 B fragments from the LDS panel + 4 WMMAs (A reused 4x) ----
#pragma unroll
        for (int nf = 0; nf < 4; ++nf) {
            bf16x16 bfr;
#pragma unroll
            for (int v = 0; v < 8; ++v) {
                const int kk = ((v < 4) ? 0 : 16) + hi * 8 + (v & 3) * 2;
                const float b0 = P[kk * 64 + nf * 16];        // ds_load_b32
                const float b1 = P[(kk + 1) * 64 + nf * 16];
                bfr[2 * v]     = (__bf16)b0;
                bfr[2 * v + 1] = (__bf16)b1;
            }
            acc[nf] = __builtin_amdgcn_wmma_f32_16x16x32_bf16(
                false, af, false, bfr, (short)0, acc[nf], false, false);
        }
        __syncthreads();   // all waves done with this panel before it is re-DMA'd
    }

    // ---- epilogue: C/D layout lane%16 == N, row = base + v + 8*(lane>>4) ----
#pragma unroll
    for (int nf = 0; nf < 4; ++nf) {
        const int n  = colBase + nf * 16 + l15;
        const float bv = bias ? bias[n] : 0.0f;
#pragma unroll
        for (int v = 0; v < 8; ++v) {
            const int r = rowBase + v + 8 * hi;
            const long long crow = (long long)(r / c_rpb) * c_obs + c_off + (r % c_rpb);
            float o = acc[nf][v] + bv;
            if (SOFTPLUS) o = (o > 20.0f) ? o : log1pf(__expf(o));
            C[crow * (long long)N + n] = o;
        }
    }
}

// ---------------------------------------------------------------------------
__global__ void gamba_add(const float* __restrict__ a, const float* __restrict__ b,
                          float* __restrict__ o, long long total)
{
    const long long i = (long long)blockIdx.x * blockDim.x + threadIdx.x;
    if (i < total) o[i] = a[i] + b[i];
}

// Broadcast three_dgs (1, n_prep, dim) into xcat[:, :n_prep, :]
__global__ void gamba_copy_dgs(const float* __restrict__ dgs, float* __restrict__ xcat,
                               int n_prep, int dim, int L, int batch)
{
    long long i = (long long)blockIdx.x * blockDim.x + threadIdx.x;
    const long long total = (long long)batch * n_prep * dim;
    if (i >= total) return;
    const int d = (int)(i % dim);
    long long r = i / dim;
    const int l  = (int)(r % n_prep);
    const int bi = (int)(r / n_prep);
    xcat[((long long)bi * L + l) * dim + d] = dgs[(long long)l * dim + d];
}

// Depthwise causal conv1d (k=4, left pad 3) over xr[..., :d_in], + SiLU
__global__ void gamba_conv_silu(const float* __restrict__ xr, const float* __restrict__ w,
                                const float* __restrict__ cb, float* __restrict__ u,
                                int L, int d_in, int n_xr, int batch)
{
    long long i = (long long)blockIdx.x * blockDim.x + threadIdx.x;
    const long long total = (long long)batch * L * d_in;
    if (i >= total) return;
    const int d = (int)(i % d_in);
    long long r = i / d_in;
    const int l  = (int)(r % L);
    const int bi = (int)(r / L);

    const float* __restrict__ base = xr + (long long)bi * L * n_xr;
    float s = cb[d];
#pragma unroll
    for (int j = 0; j < 4; ++j) {
        const int ls = l - 3 + j;
        if (ls >= 0) s += w[d * 4 + j] * base[(long long)ls * n_xr + d];
    }
    const float sig = 1.0f / (1.0f + __expf(-s));
    u[i] = s * sig;
}

// Selective scan: thread per (batch, channel d, state n); 16 states of a
// channel on 16 adjacent lanes; y = sum_n h*C via wave32 shfl-xor butterfly.
__global__ __launch_bounds__(256) void gamba_ssm_scan(
    const float* __restrict__ delta, const float* __restrict__ u,
    const float* __restrict__ xdbl,  const float* __restrict__ xr,
    const float* __restrict__ A_log, const float* __restrict__ Dvec,
    float* __restrict__ y,
    int L, int d_in, int n_state, int dt_rank, int n_xr, int batch)
{
    const int t = blockIdx.x * blockDim.x + threadIdx.x;
    const int n  = t & (n_state - 1);
    const int d  = (t / n_state) % d_in;
    const int bi =  t / (n_state * d_in);
    if (bi >= batch) return;

    const int xdw = dt_rank + 2 * n_state;           // 64
    const float Adn = -__expf(A_log[d * n_state + n]);
    const float Dd  = Dvec[d];

    float h = 0.0f;
    for (int l = 0; l < L; ++l) {
        const long long bl = (long long)bi * L + l;
        const float dv = delta[bl * d_in + d];
        const float uv = u[bl * d_in + d];
        const float Bv = xdbl[bl * xdw + dt_rank + n];
        const float Cv = xdbl[bl * xdw + dt_rank + n_state + n];

        h = __expf(dv * Adn) * h + dv * Bv * uv;     // h = dA*h + dBu
        float yv = h * Cv;
        yv += __shfl_xor(yv, 1, 32);
        yv += __shfl_xor(yv, 2, 32);
        yv += __shfl_xor(yv, 4, 32);
        yv += __shfl_xor(yv, 8, 32);

        if (n == 0) {
            const float res = xr[bl * n_xr + d_in + d];
            const float sig = 1.0f / (1.0f + __expf(-res));
            y[bl * d_in + d] = (yv + uv * Dd) * (res * sig);
        }
    }
}

// ---------------------------------------------------------------------------
extern "C" void kernel_launch(void* const* d_in, const int* in_sizes, int n_in,
                              void* d_out, int out_size, void* d_ws, size_t ws_size,
                              hipStream_t stream)
{
    (void)in_sizes; (void)n_in; (void)out_size; (void)ws_size;

    const float* img       = (const float*)d_in[0];
    const float* pose      = (const float*)d_in[1];
    const float* W_lin     = (const float*)d_in[2];
    const float* b_lin     = (const float*)d_in[3];
    const float* dgs       = (const float*)d_in[4];
    const float* in_proj_w = (const float*)d_in[5];
    const float* conv_w    = (const float*)d_in[6];
    const float* conv_b    = (const float*)d_in[7];
    const float* x_proj_w  = (const float*)d_in[8];
    const float* dt_proj_w = (const float*)d_in[9];
    const float* dt_proj_b = (const float*)d_in[10];
    const float* A_log     = (const float*)d_in[11];
    const float* Dvec      = (const float*)d_in[12];
    const float* out_proj_w= (const float*)d_in[13];
    float* out = (float*)d_out;

    // Problem constants (from the reference)
    const int B = 2, S = 512, DIM = 512, NPREP = 2048;
    const int L = NPREP + S;             // 2560
    const int DIN = 2 * DIM;             // 1024
    const int NSTATE = 16, DTRANK = 32;
    const int NXR = 2 * DIN;             // 2048
    const int XDW = DTRANK + 2 * NSTATE; // 64

    // Workspace layout (floats)
    float* ws = (float*)d_ws;
    long long off = 0;
    float* tsum = ws + off; off += (long long)B * S * DIM;   //  0.52 M
    float* xcat = ws + off; off += (long long)B * L * DIM;   //  2.62 M
    float* xr   = ws + off; off += (long long)B * L * NXR;   // 10.49 M
    float* u    = ws + off; off += (long long)B * L * DIN;   //  5.24 M
    float* xdbl = ws + off; off += (long long)B * L * XDW;   //  0.33 M
    float* dlt  = ws + off; off += (long long)B * L * DIN;   //  5.24 M
    float* yb   = ws + off; off += (long long)B * L * DIN;   //  5.24 M

    // 0) tsum = img + pose
    {
        long long total = (long long)B * S * DIM;
        gamba_add<<<dim3((unsigned)((total + 255) / 256)), dim3(256), 0, stream>>>(
            img, pose, tsum, total);
    }

    // 1) xcat[:, :NPREP, :] = broadcast(three_dgs)
    {
        long long total = (long long)B * NPREP * DIM;
        gamba_copy_dgs<<<dim3((unsigned)((total + 255) / 256)), dim3(256), 0, stream>>>(
            dgs, xcat, NPREP, DIM, L, B);
    }

    // 2) xcat[:, NPREP:, :] = tsum @ W_lin + b_lin       (1024x512)@(512x512)
    {
        const int M = B * S;
        gamba_gemm<512, 512, 512, false>
            <<<dim3(512 / 64, M / 128), dim3(256), 0, stream>>>(
            tsum, W_lin, b_lin, xcat, M, 0, 0, /*c:*/ S, L, NPREP);
    }

    // 3) xr = xcat @ in_proj_w                          (5120x512)@(512x2048)
    {
        const int M = B * L;
        gamba_gemm<2048, 512, 512, false>
            <<<dim3(2048 / 64, M / 128), dim3(256), 0, stream>>>(
            xcat, in_proj_w, nullptr, xr, M, 0, 0, M, 0, 0);
    }

    // 4) u = silu(depthwise_causal_conv(xr[..., :DIN]) + conv_b)
    {
        long long total = (long long)B * L * DIN;
        gamba_conv_silu<<<dim3((unsigned)((total + 255) / 256)), dim3(256), 0, stream>>>(
            xr, conv_w, conv_b, u, L, DIN, NXR, B);
    }

    // 5) x_dbl = u @ x_proj_w                          (5120x1024)@(1024x64)
    {
        const int M = B * L;
        gamba_gemm<64, 1024, 1024, false>
            <<<dim3(1, M / 128), dim3(256), 0, stream>>>(
            u, x_proj_w, nullptr, xdbl, M, 0, 0, M, 0, 0);
    }

    // 6) delta = softplus(x_dbl[..., :32] @ dt_proj_w + dt_proj_b)
    {
        const int M = B * L;
        gamba_gemm<1024, 32, 64, true>
            <<<dim3(1024 / 64, M / 128), dim3(256), 0, stream>>>(
            xdbl, dt_proj_w, dt_proj_b, dlt, M, 0, 0, M, 0, 0);
    }

    // 7) selective scan -> yb = (scan_y + u*D) * silu(res)
    {
        const int total = B * DIN * NSTATE;   // 32768
        gamba_ssm_scan<<<dim3(total / 256), dim3(256), 0, stream>>>(
            dlt, u, xdbl, xr, A_log, Dvec, yb, L, DIN, NSTATE, DTRANK, NXR, B);
    }

    // 8) out = yb[:, NPREP:, :] @ out_proj_w           (1024x1024)@(1024x512)
    {
        const int M = B * S;
        gamba_gemm<512, 1024, 1024, false>
            <<<dim3(512 / 64, M / 128), dim3(256), 0, stream>>>(
            yb, out_proj_w, nullptr, out, /*a:*/ S, L, NPREP, /*c:*/ M, 0, 0);
    }
}
